// TalkingHeadAttention_58153857187941
// MI455X (gfx1250) — matrix operand order repaired and checked by reading, not compiled, not run on hardware
//
#include <hip/hip_runtime.h>

// ---------------------------------------------------------------------------
// Talking-heads attention for MI455X (gfx1250), bf16 WMMA pipeline.
// B=4, S=1024, D=768, H=12, dh=64.
// Wave tile = 64x32 (4x2 WMMA); k-loop kept un-unrolled so accumulators stay
// register-pinned (no spills, no pipeline-rotate mov chains).
// ---------------------------------------------------------------------------

#define B_  4
#define S_  1024
#define D_  768
#define H_  12
#define DH_ 64
#define BS_ (B_ * S_)      // 4096
#define N3D_ (3 * D_)      // 2304

typedef __bf16 bf16x16 __attribute__((ext_vector_type(16)));
typedef float  f32x8   __attribute__((ext_vector_type(8)));

union FragU { bf16x16 v; unsigned int u[8]; };

__device__ __forceinline__ unsigned short f2bf(float f) {
  unsigned int u = __float_as_uint(f);
  u += 0x7FFFu + ((u >> 16) & 1u);   // round-to-nearest-even
  return (unsigned short)(u >> 16);
}

__device__ __forceinline__ f32x8 wmma_bf16(const FragU& a, const FragU& b, f32x8 c) {
  // (neg_a, A, neg_b, B, c_mod, C, reuse_a, reuse_b)
  return __builtin_amdgcn_wmma_f32_16x16x32_bf16(false, a.v, false, b.v,
                                                 (short)0, c, false, false);
}

// Wave-level 64x32 GEMM tile: C(64x32,f32) += A[M,K] x BT[N,K]^T, bf16 inputs.
// A row-major [m, k] stride lda; BT row-major [n, k] stride ldb; K % 32 == 0.
// Register budget: acc 64 + A frags 32 + B frag 8  ->  no scratch spills.
template <class Epi>
__device__ __forceinline__ void wave_gemm_64x32(
    const unsigned short* __restrict__ A, int lda,
    const unsigned short* __restrict__ BT, int ldb,
    int K, int m0, int n0, Epi epi)
{
  const int lane = threadIdx.x & 31;
  const int half = lane >> 4;   // 0: lanes 0-15, 1: lanes 16-31
  const int l15  = lane & 15;

  f32x8 acc[4][2];
  const f32x8 zero = {0.f, 0.f, 0.f, 0.f, 0.f, 0.f, 0.f, 0.f};
#pragma unroll
  for (int i = 0; i < 4; ++i)
#pragma unroll
    for (int j = 0; j < 2; ++j) acc[i][j] = zero;

  // per-wave base pointers (hoisted to minimize live addressing temps)
  const unsigned short* pa0 = A  + (size_t)(m0 + l15) * lda + half * 16;
  const unsigned short* pb0 = BT + (size_t)(n0 + l15) * ldb + half * 16;

  // Body already has 8 independent WMMAs + 12 b128 loads of ILP; unrolling
  // only inflates live ranges and causes accumulator spills.
#pragma clang loop unroll(disable)
  for (int k0 = 0; k0 < K; k0 += 32) {
    // A fragments (16x32): lane row = l15; VGPR j: K = 16*half + {2j, 2j+1}
    FragU a[4];
#pragma unroll
    for (int i = 0; i < 4; ++i) {
      const unsigned short* pa = pa0 + (size_t)(16 * i) * lda + k0;
      const unsigned int* p0 = (const unsigned int*)pa;
      const unsigned int* p1 = (const unsigned int*)(pa + 8);
#pragma unroll
      for (int t = 0; t < 4; ++t) { a[i].u[t] = p0[t]; a[i].u[4 + t] = p1[t]; }
    }
    if (k0 + 32 < K) {  // prefetch next k-slab toward the WGP
      __builtin_prefetch(pa0 + k0 + 32, 0, 2);
      __builtin_prefetch(pb0 + k0 + 32, 0, 2);
    }
    // B fragments (32x16): lane col n = l15; VGPR j: K = 16*half + {2j, 2j+1}
    // loaded one at a time to keep register pressure low
#pragma unroll
    for (int j = 0; j < 2; ++j) {
      FragU b;
      const unsigned int* pb =
          (const unsigned int*)(pb0 + (size_t)(16 * j) * ldb + k0);
#pragma unroll
      for (int t = 0; t < 8; ++t) b.u[t] = pb[t];
#pragma unroll
      for (int i = 0; i < 4; ++i) acc[i][j] = wmma_bf16(a[i], b, acc[i][j]);
    }
  }
  // C layout: VGPR r -> row m0+16i+8*half+r, col n0+16j+l15
#pragma unroll
  for (int i = 0; i < 4; ++i)
#pragma unroll
    for (int j = 0; j < 2; ++j)
#pragma unroll
      for (int r = 0; r < 8; ++r)
        epi(m0 + 16 * i + 8 * half + r, n0 + 16 * j + l15, acc[i][j][r]);
}

// ---------------------------------------------------------------------------
// Prep kernels: f32 -> bf16 conversion / transposed conversion
// ---------------------------------------------------------------------------
__global__ void k_convert(const float* __restrict__ in,
                          unsigned short* __restrict__ out, int n) {
  for (int i = blockIdx.x * blockDim.x + threadIdx.x; i < n;
       i += gridDim.x * blockDim.x)
    out[i] = f2bf(in[i]);
}

// W [rows, cols] f32 -> WT [cols, rows] bf16
__global__ void k_transpose_bf16(const float* __restrict__ W,
                                 unsigned short* __restrict__ WT,
                                 int rows, int cols) {
  int total = rows * cols;
  for (int i = blockIdx.x * blockDim.x + threadIdx.x; i < total;
       i += gridDim.x * blockDim.x) {
    int r = i / cols, c = i - r * cols;
    WT[(size_t)c * rows + r] = f2bf(W[(size_t)r * cols + c]);
  }
}

// ---------------------------------------------------------------------------
// 1) QKV projection: [BS, D] x [D, 3D] -> scatter q (scaled), k, vT (bf16)
//    block = 8 waves in 2(M) x 4(N) -> 128 x 128 tile. grid(18, 32)
// ---------------------------------------------------------------------------
__global__ void k_qkv(const unsigned short* __restrict__ xb,
                      const unsigned short* __restrict__ WqkvT,
                      const float* __restrict__ bqkv,
                      unsigned short* __restrict__ qb,
                      unsigned short* __restrict__ kb,
                      unsigned short* __restrict__ vT) {
  const int wave = threadIdx.x >> 5;
  const int m0 = blockIdx.y * 128 + (wave >> 2) * 64;
  const int n0 = blockIdx.x * 128 + (wave & 3) * 32;
  wave_gemm_64x32(xb, D_, WqkvT, D_, D_, m0, n0,
    [=](int row, int col, float v) {
      float val = v + bqkv[col];
      int b = row >> 10, s = row & (S_ - 1);
      int t = col / D_, rem = col - t * D_;
      int h = rem >> 6, d = rem & (DH_ - 1);
      size_t bh = (size_t)(b * H_ + h);
      if (t == 0)       qb[(bh * S_ + s) * DH_ + d] = f2bf(val * 0.125f); // dh^-0.5
      else if (t == 1)  kb[(bh * S_ + s) * DH_ + d] = f2bf(val);
      else              vT[(bh * DH_ + d) * S_ + s] = f2bf(val);
    });
}

// ---------------------------------------------------------------------------
// 2) Scores: per (b,h): [S, dh] x [S, dh]^T -> raw logits f32 [S, S]
//    grid(8, 8, 48), block tile 128 x 128
// ---------------------------------------------------------------------------
__global__ void k_scores(const unsigned short* __restrict__ qb,
                         const unsigned short* __restrict__ kb,
                         float* __restrict__ attnRaw) {
  const int bh = blockIdx.z;
  const int wave = threadIdx.x >> 5;
  const int m0 = blockIdx.y * 128 + (wave >> 2) * 64;
  const int n0 = blockIdx.x * 128 + (wave & 3) * 32;
  const unsigned short* A  = qb + (size_t)bh * S_ * DH_;
  const unsigned short* BT = kb + (size_t)bh * S_ * DH_;
  float* C = attnRaw + (size_t)bh * S_ * S_;
  wave_gemm_64x32(A, DH_, BT, DH_, DH_, m0, n0,
    [=](int row, int col, float v) { C[(size_t)row * S_ + col] = v; });
}

// ---------------------------------------------------------------------------
// 3) Talking-heads mix -> softmax -> mix. One block per (b, q-row).
//    Thread t owns columns {t, t+256, t+512, t+768} for all 12 heads.
// ---------------------------------------------------------------------------
__global__ void k_mix_softmax(const float* __restrict__ attnRaw,
                              const float* __restrict__ Wl,
                              const float* __restrict__ bl,
                              const float* __restrict__ Ww,
                              const float* __restrict__ bw,
                              unsigned short* __restrict__ attnW) {
  __shared__ float sWl[H_ * H_], sWw[H_ * H_], sbl[H_], sbw[H_];
  __shared__ float red[256];
  __shared__ float rmax[H_], rsum[H_];
  const int tid = threadIdx.x;
  if (tid < H_ * H_) { sWl[tid] = Wl[tid]; sWw[tid] = Ww[tid]; }
  if (tid < H_)      { sbl[tid] = bl[tid]; sbw[tid] = bw[tid]; }
  __syncthreads();

  const int b = blockIdx.y, q = blockIdx.x;

  float raw[H_][4];
#pragma unroll
  for (int h = 0; h < H_; ++h) {
    const float* p = attnRaw + (((size_t)(b * H_ + h)) * S_ + q) * S_;
#pragma unroll
    for (int ii = 0; ii < 4; ++ii) raw[h][ii] = p[tid + 256 * ii];
  }

  // pre-softmax head mix: mix[g] = bl[g] + sum_h raw[h] * Wl[h][g]
  float mix[H_][4];
#pragma unroll
  for (int g = 0; g < H_; ++g) {
#pragma unroll
    for (int ii = 0; ii < 4; ++ii) {
      float s = sbl[g];
#pragma unroll
      for (int h = 0; h < H_; ++h) s += raw[h][ii] * sWl[h * H_ + g];
      mix[g][ii] = s;
    }
  }

  // row max per mixed head
  for (int g = 0; g < H_; ++g) {
    float lm = fmaxf(fmaxf(mix[g][0], mix[g][1]), fmaxf(mix[g][2], mix[g][3]));
    red[tid] = lm; __syncthreads();
    for (int off = 128; off > 0; off >>= 1) {
      if (tid < off) red[tid] = fmaxf(red[tid], red[tid + off]);
      __syncthreads();
    }
    if (tid == 0) rmax[g] = red[0];
    __syncthreads();
  }
  // exp and row sum
  for (int g = 0; g < H_; ++g) {
    float m = rmax[g], ls = 0.f;
#pragma unroll
    for (int ii = 0; ii < 4; ++ii) { mix[g][ii] = __expf(mix[g][ii] - m); ls += mix[g][ii]; }
    red[tid] = ls; __syncthreads();
    for (int off = 128; off > 0; off >>= 1) {
      if (tid < off) red[tid] += red[tid + off];
      __syncthreads();
    }
    if (tid == 0) rsum[g] = red[0];
    __syncthreads();
  }
#pragma unroll
  for (int g = 0; g < H_; ++g) {
    float inv = 1.0f / rsum[g];
#pragma unroll
    for (int ii = 0; ii < 4; ++ii) mix[g][ii] *= inv;
  }

  // post-softmax head mix -> bf16 probabilities
#pragma unroll
  for (int g = 0; g < H_; ++g) {
    unsigned short* p = attnW + (((size_t)(b * H_ + g)) * S_ + q) * S_;
#pragma unroll
    for (int ii = 0; ii < 4; ++ii) {
      float s = sbw[g];
#pragma unroll
      for (int h = 0; h < H_; ++h) s += mix[h][ii] * sWw[h * H_ + g];
      p[tid + 256 * ii] = f2bf(s);
    }
  }
}

// ---------------------------------------------------------------------------
// 4) AV: per (b,h): [S, S] x [dh, S]^T -> ctx bf16 [B, S, D]
//    8 waves in 4(M) x 2(N): block tile 256 x 64. grid(4, 48)
// ---------------------------------------------------------------------------
__global__ void k_av(const unsigned short* __restrict__ attnW,
                     const unsigned short* __restrict__ vT,
                     unsigned short* __restrict__ ctx) {
  const int bh = blockIdx.y;
  const int b = bh / H_, h = bh - b * H_;
  const int wave = threadIdx.x >> 5;
  const int m0 = blockIdx.x * 256 + (wave >> 1) * 64;
  const int n0 = (wave & 1) * 32;
  const unsigned short* A  = attnW + (size_t)bh * S_ * S_;
  const unsigned short* BT = vT    + (size_t)bh * DH_ * S_;
  wave_gemm_64x32(A, S_, BT, S_, S_, m0, n0,
    [=](int row, int col, float v) {
      ctx[((size_t)b * S_ + row) * D_ + h * DH_ + col] = f2bf(v);
    });
}

// ---------------------------------------------------------------------------
// 5) Output projection: [BS, D] x [D, D] + bias -> f32 out. grid(6, 32)
// ---------------------------------------------------------------------------
__global__ void k_proj(const unsigned short* __restrict__ ctx,
                       const unsigned short* __restrict__ WprojT,
                       const float* __restrict__ bproj,
                       float* __restrict__ out) {
  const int wave = threadIdx.x >> 5;
  const int m0 = blockIdx.y * 128 + (wave >> 2) * 64;
  const int n0 = blockIdx.x * 128 + (wave & 3) * 32;
  wave_gemm_64x32(ctx, D_, WprojT, D_, D_, m0, n0,
    [=](int row, int col, float v) {
      out[(size_t)row * D_ + col] = v + bproj[col];
    });
}

// ---------------------------------------------------------------------------
extern "C" void kernel_launch(void* const* d_in, const int* in_sizes, int n_in,
                              void* d_out, int out_size, void* d_ws, size_t ws_size,
                              hipStream_t stream) {
  (void)in_sizes; (void)n_in; (void)out_size; (void)ws_size;

  const float* x     = (const float*)d_in[0];
  const float* Wqkv  = (const float*)d_in[1];
  const float* bqkv  = (const float*)d_in[2];
  const float* Wl    = (const float*)d_in[3];
  const float* bl    = (const float*)d_in[4];
  const float* Ww    = (const float*)d_in[5];
  const float* bw    = (const float*)d_in[6];
  const float* Wproj = (const float*)d_in[7];
  const float* bproj = (const float*)d_in[8];
  float* out = (float*)d_out;

  // workspace carve-up (256B aligned)
  char* p = (char*)d_ws;
  auto carve = [&](size_t bytes) {
    char* r = p; p += (bytes + 255) & ~(size_t)255; return r;
  };
  unsigned short* xb      = (unsigned short*)carve((size_t)BS_ * D_ * 2);
  unsigned short* WqkvT   = (unsigned short*)carve((size_t)N3D_ * D_ * 2);
  unsigned short* WprojT  = (unsigned short*)carve((size_t)D_ * D_ * 2);
  unsigned short* qb      = (unsigned short*)carve((size_t)B_ * H_ * S_ * DH_ * 2);
  unsigned short* kb      = (unsigned short*)carve((size_t)B_ * H_ * S_ * DH_ * 2);
  unsigned short* vT      = (unsigned short*)carve((size_t)B_ * H_ * DH_ * S_ * 2);
  float*          attnRaw = (float*)carve((size_t)B_ * H_ * S_ * S_ * 4);
  unsigned short* attnW   = (unsigned short*)carve((size_t)B_ * H_ * S_ * S_ * 2);
  unsigned short* ctx     = (unsigned short*)carve((size_t)BS_ * D_ * 2);

  // prep: bf16 conversions + weight transposes
  {
    int n = BS_ * D_;
    k_convert<<<dim3((n + 255) / 256), dim3(256), 0, stream>>>(x, xb, n);
  }
  {
    int n = D_ * N3D_;
    k_transpose_bf16<<<dim3((n + 255) / 256), dim3(256), 0, stream>>>(Wqkv, WqkvT, D_, N3D_);
  }
  {
    int n = D_ * D_;
    k_transpose_bf16<<<dim3((n + 255) / 256), dim3(256), 0, stream>>>(Wproj, WprojT, D_, D_);
  }

  // 1) QKV projection
  k_qkv<<<dim3(N3D_ / 128, BS_ / 128), dim3(256), 0, stream>>>(
      xb, WqkvT, bqkv, qb, kb, vT);

  // 2) raw attention logits
  k_scores<<<dim3(S_ / 128, S_ / 128, B_ * H_), dim3(256), 0, stream>>>(
      qb, kb, attnRaw);

  // 3) head-mix -> softmax -> head-mix
  k_mix_softmax<<<dim3(S_, B_), dim3(256), 0, stream>>>(
      attnRaw, Wl, bl, Ww, bw, attnW);

  // 4) attention-weighted values
  k_av<<<dim3(S_ / 256, B_ * H_), dim3(256), 0, stream>>>(attnW, vT, ctx);

  // 5) output projection
  k_proj<<<dim3(D_ / 128, BS_ / 128), dim3(256), 0, stream>>>(
      ctx, WprojT, bproj, out);
}
